// DCA_50594714746892
// MI455X (gfx1250) — matrix operand order repaired and checked by reading, not compiled, not run on hardware
//
#include <hip/hip_runtime.h>
#include <hip/hip_bf16.h>

// ---------------------------------------------------------------------------
// Deformable cross-attention for MI455X (gfx1250, wave32, WMMA + async LDS).
// B=8, DIM=256, H=W=48 (HW=2304), HEADS=8, DH=32, Hk=Wk=24 (NKV=576).
// Contractions: v_wmma_f32_16x16x32_f16. Streamed operands staged to LDS with
// global_load_async_to_lds_b128 (ASYNCcnt, double-buffered), then re-packed
// ONCE PER BLOCK into K-pair-per-dword f16 tiles so WMMA operands are raw
// ds_load dwords (no per-wave cvt chains). Weights pre-packed to f16 once.
// ---------------------------------------------------------------------------

typedef __attribute__((ext_vector_type(16))) _Float16 v16h;
typedef __attribute__((ext_vector_type(8)))  float    v8f;

#define DIMC 256
#define HW   2304
#define NKV  576
#define ATT_SCALE 0.17677669529663688f   // 32^-0.5

__device__ __forceinline__ v8f wmma16(v16h a, v16h b, v8f c) {
  // D = A(16x32 f16) * B(32x16 f16) + C(16x16 f32)
  return __builtin_amdgcn_wmma_f32_16x16x32_f16(
      /*neg_a=*/false, a, /*neg_b=*/false, b,
      /*c_mod=*/(short)0, c, /*reuse_a=*/false, /*reuse_b=*/false);
}

__device__ __forceinline__ v8f vzero8() {
  v8f z;
#pragma unroll
  for (int i = 0; i < 8; ++i) z[i] = 0.0f;
  return z;
}

__device__ __forceinline__ float wave_sum32(float v) {
#pragma unroll
  for (int m = 1; m < 32; m <<= 1) v += __shfl_xor(v, m, 32);
  return v;
}

// Pack two floats into one dword of two f16 (v_cvt_pk_f16_f32).
__device__ __forceinline__ unsigned pkf16(float a, float b) {
  union { _Float16 h[2]; unsigned u; } r;
  r.h[0] = (_Float16)a; r.h[1] = (_Float16)b;
  return r.u;
}

// LDS byte-offset of a generic pointer to __shared__ (addr[31:0] is the
// LDS offset per the flat-aperture rules).
__device__ __forceinline__ unsigned lds_off(const void* p) {
  return (unsigned)(unsigned long long)p;
}

// Async global -> LDS copy, 16 bytes per lane. Tracked with ASYNCcnt.
__device__ __forceinline__ void async_b128(unsigned ldsoff, const float* g) {
  asm volatile("global_load_async_to_lds_b128 %0, %1, off"
               :: "v"(ldsoff), "v"(g) : "memory");
}
__device__ __forceinline__ void wait_async_le2() {
  asm volatile("s_wait_asynccnt 0x2" ::: "memory");
}
__device__ __forceinline__ void wait_async_0() {
  asm volatile("s_wait_asynccnt 0x0" ::: "memory");
}

// ---------------------------------------------------------------------------
// One-off: pack a 256x256 fp32 weight matrix into f16-pair dwords:
// wp[m*128 + kp] = (f16(W[m][2kp]), f16(W[m][2kp+1])).  grid 128 x 256.
// ---------------------------------------------------------------------------
__global__ __launch_bounds__(256) void pack_w_f16(const float* __restrict__ w,
                                                   unsigned* __restrict__ wp) {
  const int i = blockIdx.x * 256 + threadIdx.x;    // 0 .. 32767
  wp[i] = pkf16(w[2 * i], w[2 * i + 1]);
}

// ---------------------------------------------------------------------------
// 1x1-conv GEMM:  Y[b, m, p] = sum_c W[m,c] * X[b, c, p] + bias[m]
// Block = 8 waves sharing one 32(K) x 64(N) X tile:
//   async fp32 stage (double-buffered) -> block-wide pack to f16 pairs ->
//   per-wave 8x ds_load_b32 per WMMA B operand.  A operand = packed weights
//   (two global_load_b128 per k-step).  grid = (npix/64, 2, B), block = 256.
// ---------------------------------------------------------------------------
__global__ __launch_bounds__(256) void dca_gemm256(
    const unsigned* __restrict__ Wp, const float* __restrict__ X,
    float* __restrict__ Y, const float* __restrict__ bias, int npix) {
  __shared__ alignas(16) float    ldsX[2][32 * 64];  // fp32 async staging
  __shared__ alignas(16) unsigned ldsP[2][16 * 64];  // f16 pairs [kp][n]

  const int lane = threadIdx.x & 31;
  const int wave = threadIdx.x >> 5;
  const int h = lane >> 4;          // wave half
  const int l = lane & 15;
  const int mBase = (blockIdx.y * 8 + wave) * 16;
  const int nBase = blockIdx.x * 64;
  const float* Xb = X + (size_t)blockIdx.z * DIMC * npix;
  float* Yb = Y + (size_t)blockIdx.z * DIMC * npix;

  // staging map: thread -> two b128 chunks (rows r0 and r0+16 of the tile)
  const int r0 = threadIdx.x >> 4;          // 0..15
  const int cc = (threadIdx.x & 15) << 2;   // float column 0,4,..,60
  const unsigned lb = lds_off(&ldsX[0][0]);

  // prologue: stage k-step 0 into buffer 0
  async_b128(lb + (unsigned)((r0 * 64 + cc) * 4),
             Xb + (size_t)r0 * npix + nBase + cc);
  async_b128(lb + (unsigned)(((r0 + 16) * 64 + cc) * 4),
             Xb + (size_t)(r0 + 16) * npix + nBase + cc);

  v8f acc[4];
#pragma unroll
  for (int t = 0; t < 4; ++t) acc[t] = vzero8();

  for (int ks = 0; ks < 8; ++ks) {
    const int d = ks & 1;
    if (ks + 1 < 8) {                       // prefetch next k-step
      const int k0n = (ks + 1) * 32;
      const unsigned lbn = lb + (unsigned)((d ^ 1) * 32 * 64 * 4);
      async_b128(lbn + (unsigned)((r0 * 64 + cc) * 4),
                 Xb + (size_t)(k0n + r0) * npix + nBase + cc);
      async_b128(lbn + (unsigned)(((r0 + 16) * 64 + cc) * 4),
                 Xb + (size_t)(k0n + r0 + 16) * npix + nBase + cc);
      wait_async_le2();                     // current buffer arrived
    } else {
      wait_async_0();
    }
    __syncthreads();                        // fp32 tile visible to block

    // block-wide pack: 1024 dwords, 4 per thread
    {
      const float* src = &ldsX[d][0];
      unsigned* dst = &ldsP[d][0];
#pragma unroll
      for (int r = 0; r < 4; ++r) {
        const int idx = threadIdx.x + r * 256;   // = kp*64 + n
        const int kp = idx >> 6, n = idx & 63;
        dst[idx] = pkf16(src[(2 * kp) * 64 + n], src[(2 * kp + 1) * 64 + n]);
      }
    }
    __syncthreads();                        // packed tile ready

    // A operand from pre-packed weights: elements (2j,2j+1) -> K = k0+8h+2j
    // and K = k0+16+8h+2j  ==> dwords [m][k0/2 + 4h + j] and [.. + 8 + ..]
    const uint4* wrow =
        (const uint4*)(Wp + (size_t)(mBase + l) * 128 + (ks * 16) + 4 * h);
    union { uint4 q[2]; v16h v; } A;
    A.q[0] = wrow[0];
    A.q[1] = wrow[2];

    // B tiles: dword j -> elements (2j,2j+1) -> kp = 4h+j (j<4), 8+4h+(j-4) (j>=4)
#pragma unroll
    for (int t = 0; t < 4; ++t) {
      union { unsigned u[8]; v16h v; } Bt;
#pragma unroll
      for (int j = 0; j < 8; ++j)
        Bt.u[j] = ldsP[d][(4 * h + j + (j & 4)) * 64 + t * 16 + l];
      acc[t] = wmma16(A.v, Bt.v, acc[t]);
    }
    // no trailing barrier needed: fp32 buffer d's only reader (pack) finished
    // before the packed-ready barrier, which precedes the next async issue.
  }

#pragma unroll
  for (int t = 0; t < 4; ++t) {
#pragma unroll
    for (int r = 0; r < 8; ++r) {
      const int m = mBase + r + 8 * h;      // C/D row striping
      Yb[(size_t)m * npix + nBase + t * 16 + l] = acc[t][r] + bias[m];
    }
  }
}

// ---------------------------------------------------------------------------
// Fused: depthwise 5x5 stride-2 conv on q + bias -> LayerNorm(32ch) -> GELU
//        -> offset matvec (2x32) -> position -> bilinear sample of kv.
// One wave per output point (bh, oy, ox); lane = channel (DH=32 = wave32).
// ---------------------------------------------------------------------------
__global__ __launch_bounds__(256) void dca_offset_sample(
    const float* __restrict__ q, const float* __restrict__ kv,
    const float* __restrict__ w_dw, const float* __restrict__ b_dw,
    const float* __restrict__ ln_w, const float* __restrict__ ln_b,
    const float* __restrict__ w_off, float* __restrict__ kvs) {
  const int c   = threadIdx.x & 31;                    // channel
  const int wid = blockIdx.x * 8 + (threadIdx.x >> 5); // point id
  const int bh = wid / NKV;
  const int p  = wid % NKV;
  const int oy = p / 24, ox = p % 24;
  const int b = bh >> 3, head = bh & 7;

  const float* qc = q + ((size_t)b * DIMC + head * 32 + c) * HW;
  float s = b_dw[c];
#pragma unroll
  for (int kh = 0; kh < 5; ++kh) {
    const int iy = oy * 2 - 2 + kh;
    if (iy < 0 || iy >= 48) continue;
#pragma unroll
    for (int kw = 0; kw < 5; ++kw) {
      const int ix = ox * 2 - 2 + kw;
      if (ix < 0 || ix >= 48) continue;
      s += qc[iy * 48 + ix] * w_dw[c * 25 + kh * 5 + kw];
    }
  }
  // LayerNorm over channels (cross-lane)
  const float mu  = wave_sum32(s) * (1.0f / 32.0f);
  const float d   = s - mu;
  const float var = wave_sum32(d * d) * (1.0f / 32.0f);
  float t = d * rsqrtf(var + 1e-5f) * ln_w[c] + ln_b[c];
  // exact GELU
  t = 0.5f * t * (1.0f + erff(t * 0.70710678118654752f));
  // offset = w_off @ t  (o=0 -> y component, o=1 -> x component)
  const float o0 = wave_sum32(w_off[c] * t);
  const float o1 = wave_sum32(w_off[32 + c] * t);

  const float refy = (0.5f + (float)oy) * (2.0f / 23.0f) - 1.0f;
  const float refx = (0.5f + (float)ox) * (2.0f / 23.0f) - 1.0f;
  const float py = fminf(fmaxf(o0 + refy, -1.0f), 1.0f);
  const float px = fminf(fmaxf(o1 + refx, -1.0f), 1.0f);
  const float gx = (px + 1.0f) * 0.5f * 47.0f;
  const float gy = (py + 1.0f) * 0.5f * 47.0f;
  const float x0 = floorf(gx), y0 = floorf(gy);
  const float wx = gx - x0, wy = gy - y0;
  const int x0i = min(max((int)x0, 0), 47);
  const int x1i = min(max((int)x0 + 1, 0), 47);
  const int y0i = min(max((int)y0, 0), 47);
  const int y1i = min(max((int)y0 + 1, 0), 47);

  const float* kvc = kv + ((size_t)b * DIMC + head * 32 + c) * HW;
  const float v = (1.0f - wx) * (1.0f - wy) * kvc[y0i * 48 + x0i]
                + wx * (1.0f - wy)          * kvc[y0i * 48 + x1i]
                + (1.0f - wx) * wy          * kvc[y1i * 48 + x0i]
                + wx * wy                   * kvc[y1i * 48 + x1i];
  kvs[((size_t)b * DIMC + head * 32 + c) * NKV + p] = v;
}

// ---------------------------------------------------------------------------
// Flash attention, WMMA, online softmax. One wave = 16 queries of one head.
// All 8 waves of a block share one head (144 m-tiles % 8 == 0); K/V 32-key
// tiles: async fp32 stage (double-buffered) -> block-wide pack to f16 pairs.
// S^T = K^T(16x32) x Q(32x16) per 16-key tile; exp(S^T) C-regs feed the
// P*V^T WMMA A-operand directly (C row striping == A K striping).
// ---------------------------------------------------------------------------
__global__ __launch_bounds__(256) void dca_attn(
    const float* __restrict__ q, const float* __restrict__ kbuf,
    const float* __restrict__ vbuf, float* __restrict__ outb) {
  __shared__ alignas(16) float    ldsK32[2][32 * 32];
  __shared__ alignas(16) float    ldsV32[2][32 * 32];
  __shared__ alignas(16) unsigned ldsKp[2][16 * 32]; // f16 pairs [cp][n]
  __shared__ alignas(16) unsigned ldsVp[2][32 * 16]; // f16 pairs [c][np]

  const int lane = threadIdx.x & 31;
  const int h = lane >> 4, l = lane & 15;
  const int wid = blockIdx.x * 8 + (threadIdx.x >> 5);
  const int bh = wid / 144;
  const int mBase = (wid % 144) * 16;
  const int b = bh >> 3, head = bh & 7;

  const float* qb = q    + ((size_t)b * DIMC + head * 32) * HW;
  const float* kb = kbuf + ((size_t)b * DIMC + head * 32) * NKV;
  const float* vb = vbuf + ((size_t)b * DIMC + head * 32) * NKV;

  // staging map: thread -> one b128 of K tile + one of V tile
  const int sc = threadIdx.x >> 3;          // channel 0..31
  const int sj = (threadIdx.x & 7) << 2;    // key offset 0,4,..,28
  const unsigned lk = lds_off(&ldsK32[0][0]);
  const unsigned lv = lds_off(&ldsV32[0][0]);
  const unsigned so = (unsigned)((sc * 32 + sj) * 4);

  // prologue: stage keys [0,32) into buffer 0
  async_b128(lk + so, kb + (size_t)sc * NKV + sj);
  async_b128(lv + so, vb + (size_t)sc * NKV + sj);

  // Q as fixed B operand: B[k=c][col=m], element i -> c = 8h + i (+8 for i>=8)
  v16h bq;
#pragma unroll
  for (int i = 0; i < 16; ++i)
    bq[i] = (_Float16)qb[(size_t)(8 * h + i + (i & 8)) * HW + mBase + l];

  v8f acc0 = vzero8(), acc1 = vzero8();     // O[m][c], c in 0..15 / 16..31
  float run_m = -1e30f, run_l = 0.0f;       // per-lane stats for query m = l

  for (int it = 0; it < NKV / 32; ++it) {
    const int d = it & 1;
    if (it + 1 < NKV / 32) {                // prefetch next 32 keys
      const int n0n = (it + 1) * 32;
      const unsigned bufo = (unsigned)((d ^ 1) * 32 * 32 * 4);
      async_b128(lk + bufo + so, kb + (size_t)sc * NKV + n0n + sj);
      async_b128(lv + bufo + so, vb + (size_t)sc * NKV + n0n + sj);
      wait_async_le2();
    } else {
      wait_async_0();
    }
    __syncthreads();                        // fp32 tiles visible

    // block-wide pack: K -> [cp][n] (512 dw), V -> [c][np] (512 dw); 2+2/thread
    {
      const float* K32 = &ldsK32[d][0];
      const float* V32 = &ldsV32[d][0];
      unsigned* Kp = &ldsKp[d][0];
      unsigned* Vp = &ldsVp[d][0];
#pragma unroll
      for (int r = 0; r < 2; ++r) {
        const int ik = threadIdx.x + r * 256;        // = cp*32 + n
        const int cp = ik >> 5, n = ik & 31;
        Kp[ik] = pkf16(K32[(2 * cp) * 32 + n], K32[(2 * cp + 1) * 32 + n]);
        const int cV = ik >> 4, np = ik & 15;        // = c*16 + np
        Vp[ik] = pkf16(V32[cV * 32 + 2 * np], V32[cV * 32 + 2 * np + 1]);
      }
    }
    __syncthreads();                        // packed tiles ready

    v8f s0 = vzero8(), s1 = vzero8();
#pragma unroll
    for (int sub = 0; sub < 2; ++sub) {
      const int nl = sub * 16 + l;          // local key index (A row)
      // A = K^T: elements (2j,2j+1) -> c = 8h+2j  => cp = 4h+j,
      //          elements (8+2j..)  -> c = 16+8h+2j => cp = 8+4h+j
      union { unsigned u[8]; v16h v; } AK;
#pragma unroll
      for (int j = 0; j < 4; ++j) {
        AK.u[j]     = ldsKp[d][(4 * h + j) * 32 + nl];
        AK.u[4 + j] = ldsKp[d][(8 + 4 * h + j) * 32 + nl];
      }
      if (sub == 0) s0 = wmma16(AK.v, bq, s0);
      else          s1 = wmma16(AK.v, bq, s1);
    }
    // s{0,1}[r] = S[m = l][n = 32*it + 16*sub + r + 8h]
    float mloc = -1e30f;
#pragma unroll
    for (int r = 0; r < 8; ++r) {
      s0[r] *= ATT_SCALE; s1[r] *= ATT_SCALE;
      mloc = fmaxf(mloc, fmaxf(s0[r], s1[r]));
    }
    mloc = fmaxf(mloc, __shfl_xor(mloc, 16, 32));
    const float new_m  = fmaxf(run_m, mloc);
    const float factor = __expf(run_m - new_m);
    float lsum = 0.0f;
    v16h ap;                                 // A = P, no cross-lane movement
#pragma unroll
    for (int r = 0; r < 8; ++r) {
      const float p0 = __expf(s0[r] - new_m);
      const float p1 = __expf(s1[r] - new_m);
      lsum += p0 + p1;
      ap[r]     = (_Float16)p0;
      ap[8 + r] = (_Float16)p1;
    }
    lsum += __shfl_xor(lsum, 16, 32);
    run_l = run_l * factor + lsum;
    run_m = new_m;
    // rescale accumulators: acc reg r belongs to query m = r + 8h
#pragma unroll
    for (int r = 0; r < 8; ++r) {
      const float f = __shfl(factor, r + 8 * h, 32);
      acc0[r] *= f; acc1[r] *= f;
    }
#pragma unroll
    for (int cb = 0; cb < 2; ++cb) {
      // B = V^T: dword j -> np = 4h+j (j<4), 8+4h+(j-4) (j>=4)
      const uint4* vp4 = (const uint4*)&ldsVp[d][(cb * 16 + l) * 16 + 4 * h];
      union { uint4 q2[2]; v16h v; } BV;
      BV.q2[0] = vp4[0];
      BV.q2[1] = vp4[2];
      if (cb == 0) acc0 = wmma16(ap, BV.v, acc0);
      else         acc1 = wmma16(ap, BV.v, acc1);
    }
  }

  // normalize + store: out[b, head*32 + c, m]
  float* ob = outb + ((size_t)b * DIMC + head * 32) * HW;
#pragma unroll
  for (int r = 0; r < 8; ++r) {
    const float linv = 1.0f / __shfl(run_l, r + 8 * h, 32);
    const int m = mBase + r + 8 * h;
    ob[(size_t)l * HW + m]        = acc0[r] * linv;
    ob[(size_t)(16 + l) * HW + m] = acc1[r] * linv;
  }
}

// ---------------------------------------------------------------------------
extern "C" void kernel_launch(void* const* d_in, const int* in_sizes, int n_in,
                              void* d_out, int out_size, void* d_ws, size_t ws_size,
                              hipStream_t stream) {
  const float* x    = (const float*)d_in[0];
  const float* kv   = (const float*)d_in[1];
  const float* wq   = (const float*)d_in[2];
  const float* bq   = (const float*)d_in[3];
  const float* wk   = (const float*)d_in[4];
  const float* bk   = (const float*)d_in[5];
  const float* wv   = (const float*)d_in[6];
  const float* bv   = (const float*)d_in[7];
  const float* w_dw = (const float*)d_in[8];
  const float* b_dw = (const float*)d_in[9];
  const float* ln_w = (const float*)d_in[10];
  const float* ln_b = (const float*)d_in[11];
  const float* w_off= (const float*)d_in[12];
  const float* wo   = (const float*)d_in[13];
  const float* bo   = (const float*)d_in[14];
  float* out = (float*)d_out;

  float* qbuf = (float*)d_ws;                        // 8*256*2304
  float* kvs  = qbuf + (size_t)8 * DIMC * HW;        // 8*256*576
  float* kbuf = kvs  + (size_t)8 * DIMC * NKV;       // 8*256*576
  float* vbuf = kbuf + (size_t)8 * DIMC * NKV;       // 8*256*576
  float* abuf = vbuf + (size_t)8 * DIMC * NKV;       // 8*256*2304
  unsigned* wqp = (unsigned*)(abuf + (size_t)8 * DIMC * HW); // 4 x 32768 dw
  unsigned* wkp = wqp + 32768;
  unsigned* wvp = wkp + 32768;
  unsigned* wop = wvp + 32768;

  const dim3 blk(256);
  // 0) pack weight matrices to f16 pairs
  pack_w_f16<<<dim3(128), blk, 0, stream>>>(wq, wqp);
  pack_w_f16<<<dim3(128), blk, 0, stream>>>(wk, wkp);
  pack_w_f16<<<dim3(128), blk, 0, stream>>>(wv, wvp);
  pack_w_f16<<<dim3(128), blk, 0, stream>>>(wo, wop);
  // 1) q = wq @ x + bq
  dca_gemm256<<<dim3(HW / 64, 2, 8), blk, 0, stream>>>(wqp, x, qbuf, bq, HW);
  // 2) offsets + bilinear sampling of kv -> kvs
  dca_offset_sample<<<dim3(64 * NKV / 8), blk, 0, stream>>>(
      qbuf, kv, w_dw, b_dw, ln_w, ln_b, w_off, kvs);
  // 3,4) k/v projections
  dca_gemm256<<<dim3(NKV / 64, 2, 8), blk, 0, stream>>>(wkp, kvs, kbuf, bk, NKV);
  dca_gemm256<<<dim3(NKV / 64, 2, 8), blk, 0, stream>>>(wvp, kvs, vbuf, bv, NKV);
  // 5) flash attention
  dca_attn<<<dim3(64 * (HW / 16) / 8), blk, 0, stream>>>(qbuf, kbuf, vbuf, abuf);
  // 6) out = wo @ attn_out + bo
  dca_gemm256<<<dim3(HW / 64, 2, 8), blk, 0, stream>>>(wop, abuf, out, bo, HW);
}
